// PFA_43593918054829
// MI455X (gfx1250) — matrix operand order repaired and checked by reading, not compiled, not run on hardware
//
#include <hip/hip_runtime.h>
#include <math.h>

#define H_    256
#define W_    512
#define HW_   131072
#define NB_   4
#define NPIX_ 524288
#define PI_   3.14159265358979323846

typedef __attribute__((ext_vector_type(16))) __bf16       bf16x16;
typedef __attribute__((ext_vector_type(8)))  float        f32x8;
typedef __attribute__((ext_vector_type(4)))  unsigned int u32x4;

union BV {
  bf16x16 v;
  u32x4 q[2];
  unsigned short s[16];
};

__device__ __forceinline__ unsigned short f2bf(float f) {
  unsigned int u = __float_as_uint(f);
  u += 0x7FFFu + ((u >> 16) & 1u);           // round-to-nearest-even
  return (unsigned short)(u >> 16);
}
__device__ __forceinline__ float bf2f(unsigned short h) {
  return __uint_as_float(((unsigned int)h) << 16);
}

// ---------------------------------------------------------------------------
// Weight prep: fold BN (scale into weights, bias = b - m*s), repack to bf16
// with k ordered tap-major / ci-minor: wpack[oc][tap*CiTot + ci]
// ---------------------------------------------------------------------------
__global__ void prep_w_k(const float* __restrict__ w,
                         const float* __restrict__ g, const float* __restrict__ b,
                         const float* __restrict__ m, const float* __restrict__ v,
                         unsigned short* __restrict__ wpack, float* __restrict__ bias,
                         int Co, int CiTot, int ks) {
  int Ktot = CiTot * ks * ks;
  int i = blockIdx.x * blockDim.x + threadIdx.x;
  if (i >= Co * Ktot) return;
  int oc = i / Ktot, k = i % Ktot;
  int tap = k / CiTot, ci = k % CiTot;
  int kh = tap / ks, kw = tap % ks;
  float s = g[oc] * rsqrtf(v[oc] + 1e-5f);
  float val = w[((oc * CiTot + ci) * ks + kh) * ks + kw] * s;
  wpack[oc * Ktot + k] = f2bf(val);
  if (k == 0) bias[oc] = b[oc] - m[oc] * s;
}

// ---------------------------------------------------------------------------
// NCHW fp32 -> NHWC bf16
// ---------------------------------------------------------------------------
__global__ void nchw2nhwc_k(const float* __restrict__ in, unsigned short* __restrict__ out) {
  int p = blockIdx.x * blockDim.x + threadIdx.x;       // 0..NPIX-1
  int n = p >> 17, rem = p & (HW_ - 1);
  for (int c = 0; c < 64; ++c)
    out[(size_t)p * 64 + c] = f2bf(in[(size_t)(n * 64 + c) * HW_ + rem]);
}

// ---------------------------------------------------------------------------
// cube2equi: bilinear gather (grid math in f64 to match numpy), NHWC bf16 out
// ---------------------------------------------------------------------------
__global__ void cube2equi_k(const float* __restrict__ cube, unsigned short* __restrict__ out) {
  int p = blockIdx.x * blockDim.x + threadIdx.x;       // (n, y, x)
  int n = p >> 17, rem = p & (HW_ - 1);
  int yy = rem >> 9, xx = rem & 511;
  double theta = ((double)xx - 256.0) / 256.0 * PI_;
  double phi   = ((double)yy - 128.0) / 128.0 * (PI_ * 0.5);
  double cphi = cos(phi);
  double X = cphi * cos(theta), Y = cphi * sin(theta), Z = sin(phi);
  double ax = fabs(X), ay = fabs(Y), az = fabs(Z);
  int face;
  if (ax >= ay && ax >= az)      face = (X > 0.0) ? 0 : 1;
  else if (ay >= az)             face = (Y > 0.0) ? 2 : 3;
  else                           face = (Z > 0.0) ? 4 : 5;
  double major = (face < 2) ? ax : ((face < 4) ? ay : az);
  double u = ((face < 2) ? Y : X) / major;
  double v = ((face < 4) ? Z : Y) / major;
  double px = (u + 1.0) * 0.5 * 127.0;
  double py = (v + 1.0) * 0.5 * 127.0;
  double x0 = fmin(fmax(floor(px), 0.0), 127.0);
  double x1 = fmin(x0 + 1.0, 127.0);
  double y0 = fmin(fmax(floor(py), 0.0), 127.0);
  double y1 = fmin(y0 + 1.0, 127.0);
  float wx = (float)(px - x0), wy = (float)(py - y0);
  int ix0 = (int)x0, ix1 = (int)x1, iy0 = (int)y0, iy1 = (int)y1;
  const float* fb = cube + (size_t)(n * 6 + face) * 64 * 16384;
  int o00 = iy0 * 128 + ix0, o01 = iy0 * 128 + ix1;
  int o10 = iy1 * 128 + ix0, o11 = iy1 * 128 + ix1;
  for (int c = 0; c < 64; ++c) {
    const float* fc = fb + (size_t)c * 16384;
    float top = fc[o00] * (1.f - wx) + fc[o01] * wx;
    float bot = fc[o10] * (1.f - wx) + fc[o11] * wx;
    out[(size_t)p * 64 + c] = f2bf(top * (1.f - wy) + bot * wy);
  }
}

// ---------------------------------------------------------------------------
// Implicit-GEMM conv via V_WMMA_F32_16X16X32_BF16 (wave32), specialized on
// (KS, CI0, CI1, CIB) so the K-chunk loop fully unrolls.
//   - block = 256 threads = 8 waves; covers 64 oc x 128 pixels
//   - wave  = ALL 64 oc x 16 pixels: one B fragment feeds 4 WMMAs
//     (4 accumulators = 32 VGPRs), 4x less global B traffic per WMMA
//   - weights: per-tap [64][CiTot] bf16 slice DMA'd into LDS with
//     GLOBAL_LOAD_ASYNC_TO_LDS_B128 (ASYNCcnt) then read via ds_load_b128;
//     rows padded +16B so the 16 A-row reads hit distinct banks
//   - activations NHWC bf16 stream from global (whole tensor is L2-resident)
// BN folded: acc starts at bias. Output: NHWC bf16 slice or NCHW fp32.
// ---------------------------------------------------------------------------
template <int KS, int CI0, int CI1, int CIB>
__global__ __launch_bounds__(256)
void conv_wmma_t(const unsigned short* __restrict__ in0,
                 const unsigned short* __restrict__ in1,
                 const float* __restrict__ bcast,
                 const unsigned short* __restrict__ wpack,
                 const float* __restrict__ bias,
                 unsigned short* __restrict__ outB,
                 float* __restrict__ outF,
                 int dil, int relu, int CoTot, int coOff) {
  constexpr int CITOT = CI0 + CI1 + CIB;
  constexpr int KTOT  = KS * KS * CITOT;
  constexpr int LROW  = CITOT + 8;             // pad 16B/row to spread LDS banks
  __shared__ unsigned short wlds[64 * LROW];

  const int tid  = threadIdx.x;
  const int wave = tid >> 5;
  const int lane = tid & 31;
  const int hi   = lane >> 4;                  // K-half for A/B, M-half for D
  const int ln   = lane & 15;                  // A row / B,D column

  const int pix = blockIdx.x * 128 + wave * 16 + ln;
  const int n = pix >> 17, rem = pix & (HW_ - 1);
  const int y = rem >> 9, x = rem & 511;

  f32x8 acc[4];
#pragma unroll
  for (int t = 0; t < 4; ++t)
#pragma unroll
    for (int r = 0; r < 8; ++r)
      acc[t][r] = bias[t * 16 + r + hi * 8];

  const u32x4 zero4 = {0u, 0u, 0u, 0u};

#pragma unroll 1
  for (int tap = 0; tap < KS * KS; ++tap) {
    // ---- async DMA this tap's [64][CITOT] weight slice into LDS ----
    __syncthreads();                            // previous tap's A reads done
    {
      constexpr int TOT = 64 * CITOT / 8;       // 16B transfers
#pragma unroll 1
      for (int e = tid; e < TOT; e += 256) {
        int oc = e / (CITOT / 8);
        int ci = (e % (CITOT / 8)) * 8;
        unsigned ldsOff = (unsigned)(uintptr_t)(wlds + oc * LROW + ci);
        const unsigned short* src = wpack + (size_t)oc * KTOT + tap * CITOT + ci;
        asm volatile("global_load_async_to_lds_b128 %0, %1, off"
                     :: "v"(ldsOff), "v"(src) : "memory");
      }
      asm volatile("s_wait_asynccnt 0x0" ::: "memory");
    }
    __syncthreads();

    const int dy = (tap / KS - (KS - 1) / 2) * dil;
    const int dx = (tap % KS - (KS - 1) / 2) * dil;
    const int iy = y + dy, ix = x + dx;
    const bool vd = (iy >= 0) & (iy < H_) & (ix >= 0) & (ix < W_);
    const int pIn = (n * H_ + iy) * W_ + ix;

#pragma unroll
    for (int cc = 0; cc < CITOT; cc += 32) {
      // --- B: 16 consecutive channels at this lane's pixel (shared by 4 tiles)
      BV b;
      const int cbase = cc + hi * 16;
      if (cc < CI0) {                           // folds at compile time
        if (vd) {
          const u32x4* pb = (const u32x4*)(in0 + (size_t)pIn * CI0 + cbase);
          b.q[0] = pb[0]; b.q[1] = pb[1];
        } else { b.q[0] = zero4; b.q[1] = zero4; }
      } else if (cc < CI0 + CI1) {
        if (vd) {
          const u32x4* pb = (const u32x4*)(in1 + (size_t)pIn * CI1 + (cbase - CI0));
          b.q[0] = pb[0]; b.q[1] = pb[1];
        } else { b.q[0] = zero4; b.q[1] = zero4; }
      } else {
        const float* p0 = bcast + n * 64 + (cbase - CI0 - CI1);
#pragma unroll
        for (int e = 0; e < 16; ++e) b.s[e] = f2bf(p0[e]);
      }
      // --- 4 oc-tiles: A from LDS, 4 independent WMMAs share this B
#pragma unroll
      for (int t = 0; t < 4; ++t) {
        BV a;
        const unsigned short* wA = wlds + (t * 16 + ln) * LROW + hi * 16 + cc;
        a.q[0] = *(const u32x4*)(wA);
        a.q[1] = *(const u32x4*)(wA + 8);
        acc[t] = __builtin_amdgcn_wmma_f32_16x16x32_bf16(
            false, a.v, false, b.v, (short)0, acc[t], false, false);
      }
    }
  }

  if (outF) {
#pragma unroll
    for (int t = 0; t < 4; ++t)
#pragma unroll
      for (int r = 0; r < 8; ++r) {
        int oc = t * 16 + r + hi * 8;
        float vv = acc[t][r];
        if (relu) vv = fmaxf(vv, 0.f);
        outF[(size_t)(n * 64 + oc) * HW_ + rem] = vv;
      }
  } else {
    size_t ob = (size_t)pix * CoTot + coOff;
#pragma unroll
    for (int t = 0; t < 4; ++t)
#pragma unroll
      for (int r = 0; r < 8; ++r) {
        int oc = t * 16 + r + hi * 8;
        float vv = acc[t][r];
        if (relu) vv = fmaxf(vv, 0.f);
        outB[ob + oc] = f2bf(vv);
      }
  }
}

// ---------------------------------------------------------------------------
// Per-(n,c) mean (and optionally max) over pixels of an NHWC bf16 tensor.
// ---------------------------------------------------------------------------
__global__ void reduce_nc_k(const unsigned short* __restrict__ in,
                            float* __restrict__ avg, float* __restrict__ mx, int doMax) {
  int n = blockIdx.x >> 6, c = blockIdx.x & 63;
  const unsigned short* p = in + (size_t)n * HW_ * 64 + c;
  float s = 0.f, m = -1e30f;
  for (int i = threadIdx.x; i < HW_; i += 256) {
    float vv = bf2f(p[(size_t)i * 64]);
    s += vv; m = fmaxf(m, vv);
  }
  __shared__ float ls[256], lm[256];
  int t = threadIdx.x;
  ls[t] = s; lm[t] = m;
  __syncthreads();
  for (int st = 128; st > 0; st >>= 1) {
    if (t < st) { ls[t] += ls[t + st]; lm[t] = fmaxf(lm[t], lm[t + st]); }
    __syncthreads();
  }
  if (t == 0) {
    avg[blockIdx.x] = ls[0] / (float)HW_;
    if (doMax) mx[blockIdx.x] = lm[0];
  }
}

// msf global-pool branch: gp5 = relu(bn(gp_w · gp))  (per n: 64x64 matvec)
__global__ void gp5_k(const float* __restrict__ gpw,
                      const float* __restrict__ g, const float* __restrict__ b,
                      const float* __restrict__ m, const float* __restrict__ v,
                      const float* __restrict__ gp, float* __restrict__ gp5) {
  int t = threadIdx.x;               // 256 = (n, co)
  int n = t >> 6, co = t & 63;
  float val = 0.f;
  for (int ci = 0; ci < 64; ++ci) val += gpw[co * 64 + ci] * gp[n * 64 + ci];
  float s = g[co] * rsqrtf(v[co] + 1e-5f);
  val = val * s + (b[co] - m[co] * s);
  gp5[t] = fmaxf(val, 0.f);
}

// channel attention: catt = sigmoid(bn(sel · [fc2(relu(fc1(avg))); fc2(relu(fc1(max)))]))
__global__ void ca_k(const float* __restrict__ fc1, const float* __restrict__ fc2,
                     const float* __restrict__ sel,
                     const float* __restrict__ g, const float* __restrict__ b,
                     const float* __restrict__ m, const float* __restrict__ v,
                     const float* __restrict__ cav, const float* __restrict__ cmx,
                     float* __restrict__ catt) {
  int t = threadIdx.x;               // 256 = (n, co)
  int n = t >> 6, co = t & 63;
  float ha[4], hm[4];
#pragma unroll
  for (int j = 0; j < 4; ++j) {
    float s1 = 0.f, s2 = 0.f;
    for (int ci = 0; ci < 64; ++ci) {
      float w = fc1[j * 64 + ci];
      s1 += w * cav[n * 64 + ci];
      s2 += w * cmx[n * 64 + ci];
    }
    ha[j] = fmaxf(s1, 0.f);
    hm[j] = fmaxf(s2, 0.f);
  }
  float val = 0.f;
#pragma unroll
  for (int j = 0; j < 4; ++j) {
    float k0 = 0.f, k1 = 0.f;
    for (int ci = 0; ci < 64; ++ci) {
      float f2 = fc2[ci * 4 + j];
      k0 += sel[co * 128 + ci] * f2;
      k1 += sel[co * 128 + 64 + ci] * f2;
    }
    val += ha[j] * k0 + hm[j] * k1;
  }
  float s = g[co] * rsqrtf(v[co] + 1e-5f);
  val = val * s + (b[co] - m[co] * s);
  catt[t] = 1.f / (1.f + expf(-val));
}

// spatial attention: softmax over 2 channels of bn(conv1x1(ms)) -> s0,s1 per pixel
__global__ void sa_k(const unsigned short* __restrict__ ms,
                     const float* __restrict__ saw,
                     const float* __restrict__ g, const float* __restrict__ b,
                     const float* __restrict__ m, const float* __restrict__ v,
                     float* __restrict__ s0, float* __restrict__ s1) {
  int p = blockIdx.x * blockDim.x + threadIdx.x;
  const unsigned short* row = ms + (size_t)p * 64;
  float d0 = 0.f, d1 = 0.f;
  for (int ci = 0; ci < 64; ++ci) {
    float vv = bf2f(row[ci]);
    d0 += saw[ci] * vv;
    d1 += saw[64 + ci] * vv;
  }
  float sc0 = g[0] * rsqrtf(v[0] + 1e-5f);
  float sc1 = g[1] * rsqrtf(v[1] + 1e-5f);
  d0 = d0 * sc0 + (b[0] - m[0] * sc0);
  d1 = d1 * sc1 + (b[1] - m[1] * sc1);
  float mm = fmaxf(d0, d1);
  float e0 = expf(d0 - mm), e1 = expf(d1 - mm);
  float inv = 1.f / (e0 + e1);
  s0[p] = e0 * inv;
  s1[p] = e1 * inv;
}

// gated inputs for the second fs: tmpA = catt*equi*s0, tmpB = catt*proj*s1
__global__ void gate_k(const unsigned short* __restrict__ ebuf,
                       const unsigned short* __restrict__ proj,
                       const float* __restrict__ catt,
                       const float* __restrict__ s0, const float* __restrict__ s1,
                       unsigned short* __restrict__ tA, unsigned short* __restrict__ tB) {
  int p = blockIdx.x * blockDim.x + threadIdx.x;
  int n = p >> 17;
  float w0 = s0[p], w1 = s1[p];
  size_t base = (size_t)p * 64;
  for (int c = 0; c < 64; ++c) {
    float ca = catt[n * 64 + c];
    tA[base + c] = f2bf(ca * bf2f(ebuf[base + c]) * w0);
    tB[base + c] = f2bf(ca * bf2f(proj[base + c]) * w1);
  }
}

// ---------------------------------------------------------------------------
extern "C" void kernel_launch(void* const* d_in, const int* in_sizes, int n_in,
                              void* d_out, int out_size, void* d_ws, size_t ws_size,
                              hipStream_t stream) {
  // --- input index tables (insertion-order vs pytree-sorted flattening) ---
  int I_equi, I_cube, I_fusw, I_fusbn, I_fsw, I_fsbn, I_a1w, I_a1bn, I_a2w, I_a2bn,
      I_a3w, I_a3bn, I_a4w, I_a4bn, I_gpw, I_gpbn, I_fcw, I_fcbn, I_fc1, I_fc2,
      I_selw, I_selbn, I_saw, I_sabn;
  if (in_sizes[0] == 25165824) {  // cube first => sorted-key pytree flattening
    I_cube = 0;  I_equi = 1;  I_fc1 = 2;  I_fc2 = 3;  I_selbn = 4;  I_selw = 8;
    I_fsbn = 9;  I_fsw = 13;  I_fusbn = 14; I_fusw = 18;
    I_a1bn = 19; I_a1w = 23;  I_a2bn = 24; I_a2w = 28;
    I_a3bn = 29; I_a3w = 33;  I_a4bn = 34; I_a4w = 38;
    I_fcbn = 39; I_fcw = 43;  I_gpbn = 44; I_gpw = 48;
    I_sabn = 49; I_saw = 53;
  } else {                        // insertion-order flattening
    I_equi = 0;  I_cube = 1;  I_fusw = 2;  I_fusbn = 3;
    I_fsw = 7;   I_fsbn = 8;  I_a1w = 12;  I_a1bn = 13;
    I_a2w = 17;  I_a2bn = 18; I_a3w = 22;  I_a3bn = 23;
    I_a4w = 27;  I_a4bn = 28; I_gpw = 32;  I_gpbn = 33;
    I_fcw = 37;  I_fcbn = 38; I_fc1 = 42;  I_fc2 = 43;
    I_selw = 44; I_selbn = 45; I_saw = 49; I_sabn = 50;
  }
  auto F = [&](int i) -> const float* { return (const float*)d_in[i]; };

  // --- workspace bump allocator ---
  char* ws = (char*)d_ws;
  size_t off = 0;
  auto alloc = [&](size_t bytes) -> char* {
    char* p = ws + off;
    off += (bytes + 255) & ~(size_t)255;
    return p;
  };
  unsigned short* ebuf  = (unsigned short*)alloc((size_t)NPIX_ * 64 * 2); // equi NHWC bf16
  unsigned short* c2eb  = (unsigned short*)alloc((size_t)NPIX_ * 64 * 2); // cube2equi / tmpA
  unsigned short* proj  = (unsigned short*)alloc((size_t)NPIX_ * 64 * 2);
  unsigned short* xbuf  = (unsigned short*)alloc((size_t)NPIX_ * 64 * 2); // msf input (both passes)
  unsigned short* big   = (unsigned short*)alloc((size_t)NPIX_ * 256 * 2); // branch slices / tmpB
  unsigned short* msbuf = (unsigned short*)alloc((size_t)NPIX_ * 64 * 2);
  unsigned short* fused = (unsigned short*)alloc((size_t)NPIX_ * 64 * 2);
  float* s0   = (float*)alloc((size_t)NPIX_ * 4);
  float* s1   = (float*)alloc((size_t)NPIX_ * 4);
  float* gp   = (float*)alloc(1024);
  float* gp5  = (float*)alloc(1024);
  float* catt = (float*)alloc(1024);
  float* cav  = (float*)alloc(1024);
  float* cmx  = (float*)alloc(1024);
  unsigned short* wpFus = (unsigned short*)alloc(64 * 576 * 2);
  unsigned short* wpFs  = (unsigned short*)alloc(64 * 1152 * 2);
  unsigned short* wpA1  = (unsigned short*)alloc(64 * 64 * 2);
  unsigned short* wpA2  = (unsigned short*)alloc(64 * 576 * 2);
  unsigned short* wpA3  = (unsigned short*)alloc(64 * 576 * 2);
  unsigned short* wpA4  = (unsigned short*)alloc(64 * 576 * 2);
  unsigned short* wpFc  = (unsigned short*)alloc(64 * 320 * 2);
  float* bFus = (float*)alloc(256); float* bFs = (float*)alloc(256);
  float* bA1  = (float*)alloc(256); float* bA2 = (float*)alloc(256);
  float* bA3  = (float*)alloc(256); float* bA4 = (float*)alloc(256);
  float* bFc  = (float*)alloc(256);
  (void)ws_size; (void)n_in; (void)out_size;

  // --- weight prep (BN-folded, bf16, tap-major K) ---
  auto prep = [&](int iw, int ibn, unsigned short* wp, float* bs, int Ci, int ks) {
    int thr = 64 * Ci * ks * ks;
    prep_w_k<<<dim3((thr + 255) / 256), dim3(256), 0, stream>>>(
        F(iw), F(ibn), F(ibn + 1), F(ibn + 2), F(ibn + 3), wp, bs, 64, Ci, ks);
  };
  prep(I_fusw, I_fusbn, wpFus, bFus, 64, 3);
  prep(I_fsw,  I_fsbn,  wpFs,  bFs, 128, 3);
  prep(I_a1w,  I_a1bn,  wpA1,  bA1,  64, 1);
  prep(I_a2w,  I_a2bn,  wpA2,  bA2,  64, 3);
  prep(I_a3w,  I_a3bn,  wpA3,  bA3,  64, 3);
  prep(I_a4w,  I_a4bn,  wpA4,  bA4,  64, 3);
  prep(I_fcw,  I_fcbn,  wpFc,  bFc, 320, 1);

  dim3 gPix(NPIX_ / 256), b256(256);
  dim3 gConv(NPIX_ / 128);

  // conv wrappers (template-specialized kernels)
  auto conv3 = [&](const unsigned short* in0, const unsigned short* wp, const float* bs,
                   unsigned short* oB, float* oF, int dil, int relu, int CoTot, int coOff) {
    conv_wmma_t<3, 64, 0, 0><<<gConv, b256, 0, stream>>>(
        in0, nullptr, nullptr, wp, bs, oB, oF, dil, relu, CoTot, coOff);
  };
  auto convFs = [&](const unsigned short* in0, const unsigned short* in1,
                    unsigned short* oB, float* oF) {
    conv_wmma_t<3, 64, 64, 0><<<gConv, b256, 0, stream>>>(
        in0, in1, nullptr, wpFs, bFs, oB, oF, 1, 0, 64, 0);
  };
  auto msf = [&](unsigned short* xin, unsigned short* xout) {
    conv_wmma_t<1, 64, 0, 0><<<gConv, b256, 0, stream>>>(
        xin, nullptr, nullptr, wpA1, bA1, big, nullptr, 1, 1, 256, 0);
    conv3(xin, wpA2, bA2, big, nullptr, 6,  1, 256,  64);
    conv3(xin, wpA3, bA3, big, nullptr, 12, 1, 256, 128);
    conv3(xin, wpA4, bA4, big, nullptr, 18, 1, 256, 192);
    reduce_nc_k<<<dim3(256), b256, 0, stream>>>(xin, gp, cav, 0);
    gp5_k<<<dim3(1), b256, 0, stream>>>(F(I_gpw), F(I_gpbn), F(I_gpbn + 1),
                                        F(I_gpbn + 2), F(I_gpbn + 3), gp, gp5);
    conv_wmma_t<1, 256, 0, 64><<<gConv, b256, 0, stream>>>(
        big, nullptr, gp5, wpFc, bFc, xout, nullptr, 1, 1, 64, 0);
  };

  // 1) layout transforms + cube2equi projection
  nchw2nhwc_k<<<gPix, b256, 0, stream>>>(F(I_equi), ebuf);
  cube2equi_k<<<gPix, b256, 0, stream>>>(F(I_cube), c2eb);
  // 2) proj = bn(conv3x3(c2e))
  conv3(c2eb, wpFus, bFus, proj, nullptr, 1, 0, 64, 0);
  // 3) msin = fs(equi, proj)
  convFs(ebuf, proj, xbuf, nullptr);
  // 4) ms = msf(msin)
  msf(xbuf, msbuf);
  // 5) channel attention on ms
  reduce_nc_k<<<dim3(256), b256, 0, stream>>>(msbuf, cav, cmx, 1);
  ca_k<<<dim3(1), b256, 0, stream>>>(F(I_fc1), F(I_fc2), F(I_selw),
                                     F(I_selbn), F(I_selbn + 1), F(I_selbn + 2),
                                     F(I_selbn + 3), cav, cmx, catt);
  // 6) spatial attention softmax on ms
  sa_k<<<gPix, b256, 0, stream>>>(msbuf, F(I_saw), F(I_sabn), F(I_sabn + 1),
                                  F(I_sabn + 2), F(I_sabn + 3), s0, s1);
  // 7) gated inputs: tmpA = catt*equi*s0 (reuse c2eb), tmpB = catt*proj*s1 (head of big)
  gate_k<<<gPix, b256, 0, stream>>>(ebuf, proj, catt, s0, s1, c2eb, big);
  // 8) fused = msf(fs(tmpA, tmpB))
  convFs(c2eb, big, xbuf, nullptr);
  msf(xbuf, fused);
  // 9) out = fs(ms, fused) -> fp32 NCHW directly into d_out
  convFs(msbuf, fused, nullptr, (float*)d_out);
}
